// PoseCorrectionNetwork_22368189677586
// MI455X (gfx1250) — matrix operand order repaired and checked by reading, not compiled, not run on hardware
//
#include <hip/hip_runtime.h>

// ---------------------------------------------------------------------------
// DGCNN pose-correction net for MI455X (gfx1250, wave32, WMMA).
// Edge-conv rewritten as  y[o,n,k] = U[o, idx[n,k]] + V[o,n]  with
//   U = W[:, :C] @ X,  V = (W[:, C:] - W[:, :C]) @ X
// computed by stacked bf16 WMMA GEMMs (f32 accumulate). conv5 stages its
// B panel into LDS with double-buffered global_load_async_to_lds_b128.
// ---------------------------------------------------------------------------

typedef __bf16 bf16_t;
typedef __attribute__((ext_vector_type(16))) __bf16 v16bf;
typedef __attribute__((ext_vector_type(8)))  __bf16 v8bf;
typedef __attribute__((ext_vector_type(8)))  float  v8f;

#define BNS_  0.9999950000374997f   // 1/sqrt(1+1e-5)
#define NPTS  1024
#define KNN_K 20
#define NBATCH 16
#define HCH   512                    // 64+64+128+256 concat channels

static __device__ __forceinline__ bf16_t f2bf(float f) {
    union { float f; unsigned u; } v; v.f = f;
    unsigned u = v.u;
    u += 0x7fffu + ((u >> 16) & 1u);          // round-to-nearest-even
    union { unsigned short s; bf16_t b; } o; o.s = (unsigned short)(u >> 16);
    return o.b;
}

// ---------------------------------------------------------------------------
// KNN: one wave per point. d[i] = -||x_n - x_m||^2 for m = lane + 32*i.
// 20 rounds of (local unrolled argmax) + (cross-lane shfl_xor argmax).
// ---------------------------------------------------------------------------
__global__ __launch_bounds__(32)
void knn_kernel(const float* __restrict__ X, int C, int batchStride,
                int* __restrict__ idx) {
    int bn = blockIdx.x;
    int b = bn >> 10, n = bn & (NPTS - 1);
    int lane = threadIdx.x;
    const float* xb = X + (size_t)b * batchStride;

    float d[32];
#pragma unroll
    for (int i = 0; i < 32; ++i) d[i] = 0.0f;

    for (int c = 0; c < C; ++c) {
        const float* row = xb + c * NPTS;
        float cen = row[n];
#pragma unroll
        for (int i = 0; i < 32; ++i) {
            float t = row[lane + 32 * i] - cen;
            d[i] = fmaf(-t, t, d[i]);
        }
    }

    for (int r = 0; r < KNN_K; ++r) {
        float bv = -3.0e38f; int bi = 0;
#pragma unroll
        for (int i = 0; i < 32; ++i) {
            bool g = d[i] > bv;
            bv = g ? d[i] : bv;
            bi = g ? (lane + 32 * i) : bi;
        }
#pragma unroll
        for (int off = 16; off >= 1; off >>= 1) {
            float ov = __shfl_xor(bv, off, 32);
            int   oi = __shfl_xor(bi, off, 32);
            if (ov > bv || (ov == bv && oi < bi)) { bv = ov; bi = oi; }
        }
        if (lane == 0) idx[(size_t)bn * KNN_K + r] = bi;
        if ((bi & 31) == lane) {
            int p = bi >> 5;
#pragma unroll
            for (int i = 0; i < 32; ++i) if (i == p) d[i] = -3.4e38f;
        }
    }
}

// ---------------------------------------------------------------------------
// Convert [b][C][n] f32 activations -> [b][n][Cpad] bf16 (transposed, padded).
// ---------------------------------------------------------------------------
__global__ void cvt_act_kernel(const float* __restrict__ in, int batchStride,
                               int Ci, int Cpad, bf16_t* __restrict__ out) {
    int bn = blockIdx.x;
    int b = bn >> 10, n = bn & (NPTS - 1);
    int c = threadIdx.x;
    float v = (c < Ci) ? in[(size_t)b * batchStride + c * NPTS + n] : 0.0f;
    out[((size_t)bn) * Cpad + c] = f2bf(v);
}

// Build stacked bf16 weights: rows [0,O)   = W[:, :Ci]            (feat part)
//                             rows [O,2O)  = W[:, Ci:] - W[:, :Ci] (center)
__global__ void wstack_kernel(const float* __restrict__ W, int Ci, int O,
                              int Cpad, bf16_t* __restrict__ out) {
    int row = blockIdx.x, c = threadIdx.x;
    float v = 0.0f;
    if (c < Ci) {
        if (row < O)  v = W[(size_t)row * (2 * Ci) + c];
        else {
            int o = row - O;
            v = W[(size_t)o * (2 * Ci) + Ci + c] - W[(size_t)o * (2 * Ci) + c];
        }
    }
    out[(size_t)row * Cpad + c] = f2bf(v);
}

__global__ void cvt_plain_kernel(const float* __restrict__ in, int n,
                                 bf16_t* __restrict__ out) {
    int i = blockIdx.x * blockDim.x + threadIdx.x;
    if (i < n) out[i] = f2bf(in[i]);
}

// ---------------------------------------------------------------------------
// bf16 WMMA GEMM, register-blocked 16x64 per wave (4 n-tiles share one A
// fragment). D[b][m][n] = A[m][Kd] * B[b][n][Kd]^T. Fragment element ->
// matrix mapping uses contiguous k = 16*khalf + s in both A and B.
// ---------------------------------------------------------------------------
__global__ __launch_bounds__(32)
void gemm_bf16_kernel(const bf16_t* __restrict__ A,   // [M][Kd] row-major
                      const bf16_t* __restrict__ Bm,  // [b][N][Kd]
                      float* __restrict__ D,          // [b][512][N]
                      int Kd) {
    int nt0 = blockIdx.x * 4, mt = blockIdx.y, b = blockIdx.z;
    int lane = threadIdx.x;
    int ml = lane & 15, khalf = lane >> 4;

    v8f acc[4] = {{}, {}, {}, {}};
    const bf16_t* arow = A  + ((size_t)(mt * 16 + ml)) * Kd + khalf * 16;
    const bf16_t* brow = Bm + ((size_t)b * NPTS + nt0 * 16 + ml) * Kd + khalf * 16;
    const size_t tstride = (size_t)16 * Kd;

    for (int kk = 0; kk < Kd; kk += 32) {
        v8bf a0 = *(const v8bf*)(arow + kk);        // A: k = kk + 16*khalf + [0,8)
        v8bf a1 = *(const v8bf*)(arow + kk + 8);    //    k = kk + 16*khalf + 8 + [0,8)
        v16bf av = __builtin_shufflevector(a0, a1, 0,1,2,3,4,5,6,7,
                                                   8,9,10,11,12,13,14,15);
#pragma unroll
        for (int t = 0; t < 4; ++t) {
            v16bf bv = *(const v16bf*)(brow + t * tstride + kk); // k = kk+16*khalf+[0,16)
            acc[t] = __builtin_amdgcn_wmma_f32_16x16x32_bf16(
                         false, av, false, bv, (short)0, acc[t], false, false);
        }
    }

#pragma unroll
    for (int t = 0; t < 4; ++t) {
        float* drow = D + ((size_t)b * HCH + mt * 16 + 8 * khalf) * NPTS
                        + (nt0 + t) * 16 + ml;
#pragma unroll
        for (int j = 0; j < 8; ++j) drow[(size_t)j * NPTS] = acc[t][j];
    }
}

// ---------------------------------------------------------------------------
// Gather + BN + leakyReLU + max over k neighbors.
// out[b][chanOff+o][n] = max_k lrelu(g*(BNS*(U[o,idx]+V[o,n])) + bias)
// ---------------------------------------------------------------------------
__global__ void gathermax_kernel(const float* __restrict__ Ust,
                                 const int* __restrict__ idx,
                                 const float* __restrict__ g,
                                 const float* __restrict__ bia,
                                 float* __restrict__ Hout,
                                 int O, int chanOff) {
    int bn = blockIdx.x;
    int b = bn >> 10, n = bn & (NPTS - 1);
    __shared__ int s[KNN_K];
    if (threadIdx.x < KNN_K) s[threadIdx.x] = idx[(size_t)bn * KNN_K + threadIdx.x];
    __syncthreads();

    int o = threadIdx.x;
    const float* Ub = Ust + (size_t)b * HCH * NPTS;
    float vcen = Ub[(size_t)(O + o) * NPTS + n];
    float gg = g[o] * BNS_, bb = bia[o];
    float best = -3.4e38f;
#pragma unroll
    for (int k = 0; k < KNN_K; ++k) {
        float r = Ub[(size_t)o * NPTS + s[k]] + vcen;
        float y = fmaf(gg, r, bb);
        y = (y > 0.0f) ? y : 0.2f * y;
        best = fmaxf(best, y);
    }
    Hout[((size_t)b * HCH + chanOff + o) * NPTS + n] = best;
}

// ---------------------------------------------------------------------------
// conv5 (1024x512x1024 GEMM) fused with BN+leaky and max/mean pooling.
// 4 waves per block own 4 m-tiles; the 16-row B tile (contiguous 16KB in the
// [n][c] layout) is double-buffer staged into LDS with
// global_load_async_to_lds_b128 (ASYNCcnt, in-order completion); A tiles are
// preloaded into registers so the inner loop is ds_load_b128 + v_wmma.
// ---------------------------------------------------------------------------
__global__ __launch_bounds__(128)
void conv5_pool_kernel(const bf16_t* __restrict__ W5,   // [1024][512]
                       const bf16_t* __restrict__ Hbf,  // [b][1024][512]
                       const float* __restrict__ g5,
                       const float* __restrict__ b5,
                       float* __restrict__ p) {         // [b][2048]
    __shared__ __align__(16) bf16_t smem[2][16 * HCH];  // 2 x 16KB tiles

    int b = blockIdx.y;
    int tid = threadIdx.x;
    int wave = tid >> 5, lane = tid & 31;
    int mt = blockIdx.x * 4 + wave;
    int ml = lane & 15, khalf = lane >> 4;

    const bf16_t* hb = Hbf + (size_t)b * NPTS * HCH;

    // stage one 16-row (16KB, contiguous) B tile into smem[buf] asynchronously:
    // each of the 128 threads copies 8 x 16B. Per-wave ASYNCcnt += 8.
    auto stage = [&](int nt, int buf) {
        unsigned l0 = (unsigned)(size_t)(&smem[buf][0]) + (unsigned)(tid * 16);
        unsigned long long g0 =
            (unsigned long long)(size_t)(hb + (size_t)nt * 16 * HCH) + (size_t)tid * 16;
#pragma unroll
        for (int j = 0; j < 8; ++j) {
            unsigned laddr = l0 + j * 2048;
            unsigned long long gaddr = g0 + (unsigned long long)j * 2048;
            asm volatile("global_load_async_to_lds_b128 %0, %1, off"
                         :: "v"(laddr), "v"(gaddr) : "memory");
        }
    };

    // preload this wave's A tile (16 rows x 512) as 16 WMMA fragments
    const bf16_t* arow = W5 + ((size_t)(mt * 16 + ml)) * HCH + khalf * 16;
    v16bf areg[16];
#pragma unroll
    for (int s = 0; s < 16; ++s) {
        v8bf a0 = *(const v8bf*)(arow + s * 32);
        v8bf a1 = *(const v8bf*)(arow + s * 32 + 8);
        areg[s] = __builtin_shufflevector(a0, a1, 0,1,2,3,4,5,6,7,
                                                  8,9,10,11,12,13,14,15);
    }

    float gv[8], bv[8], rmax[8], rsum[8];
#pragma unroll
    for (int j = 0; j < 8; ++j) {
        int row = mt * 16 + j + 8 * khalf;
        gv[j] = g5[row] * BNS_; bv[j] = b5[row];
        rmax[j] = -3.4e38f; rsum[j] = 0.0f;
    }

    stage(0, 0);
    for (int nt = 0; nt < NPTS / 16; ++nt) {
        int buf = nt & 1;
        if (nt + 1 < NPTS / 16) {
            stage(nt + 1, buf ^ 1);                 // prefetch next tile
            asm volatile("s_wait_asynccnt 0x8" ::: "memory"); // this tile done
        } else {
            asm volatile("s_wait_asynccnt 0x0" ::: "memory");
        }
        __syncthreads();                            // all waves' quarters in

        const bf16_t* sp = &smem[buf][(size_t)ml * HCH + khalf * 16];
        v8f acc = {};
#pragma unroll
        for (int s = 0; s < 16; ++s) {
            v16bf bvf = *(const v16bf*)(sp + s * 32);
            acc = __builtin_amdgcn_wmma_f32_16x16x32_bf16(
                      false, areg[s], false, bvf, (short)0, acc, false, false);
        }
#pragma unroll
        for (int j = 0; j < 8; ++j) {
            float y = fmaf(gv[j], acc[j], bv[j]);
            y = (y > 0.0f) ? y : 0.2f * y;
            rmax[j] = fmaxf(rmax[j], y);
            rsum[j] += y;
        }
        __syncthreads();                            // buf safe to overwrite
    }

    // reduce across the 16 lanes of each half (rows differ between halves)
#pragma unroll
    for (int off = 1; off <= 8; off <<= 1) {
#pragma unroll
        for (int j = 0; j < 8; ++j) {
            rmax[j] = fmaxf(rmax[j], __shfl_xor(rmax[j], off, 32));
            rsum[j] += __shfl_xor(rsum[j], off, 32);
        }
    }
    if ((lane & 15) == 0) {
#pragma unroll
        for (int j = 0; j < 8; ++j) {
            int row = mt * 16 + j + 8 * khalf;
            p[(size_t)b * 2048 + row]        = rmax[j];
            p[(size_t)b * 2048 + 1024 + row] = rsum[j] * (1.0f / NPTS);
        }
    }
}

// ---------------------------------------------------------------------------
// Tiny MLP head: out[b][o] = act(g*(BNS*dot(in[b], W[o])) + bias)
// mode 0: BN+leaky, 1: BN+GELU(exact), 2: plain linear.
// ---------------------------------------------------------------------------
__global__ void dense_kernel(const float* __restrict__ in,
                             const float* __restrict__ W,
                             const float* __restrict__ g,
                             const float* __restrict__ bia,
                             float* __restrict__ out,
                             int I, int O, int mode) {
    int tid = blockIdx.x * blockDim.x + threadIdx.x;
    if (tid >= NBATCH * O) return;
    int b = tid / O, o = tid % O;
    const float* ir = in + (size_t)b * I;
    const float* wr = W + (size_t)o * I;
    float acc = 0.0f;
    for (int i = 0; i < I; ++i) acc = fmaf(ir[i], wr[i], acc);
    float y;
    if (mode == 0) {
        y = fmaf(g[o] * BNS_, acc, bia[o]);
        y = (y > 0.0f) ? y : 0.2f * y;
    } else if (mode == 1) {
        y = fmaf(g[o] * BNS_, acc, bia[o]);
        y = 0.5f * y * (1.0f + erff(y * 0.70710678118654752f));
    } else {
        y = acc;
    }
    out[tid] = y;
}

// ---------------------------------------------------------------------------
extern "C" void kernel_launch(void* const* d_in, const int* in_sizes, int n_in,
                              void* d_out, int out_size, void* d_ws, size_t ws_size,
                              hipStream_t stream) {
    (void)in_sizes; (void)n_in; (void)out_size; (void)ws_size;
    const float* x    = (const float*)d_in[0];
    const float* w1   = (const float*)d_in[2];
    const float* g1   = (const float*)d_in[3];
    const float* b1   = (const float*)d_in[4];
    const float* w2   = (const float*)d_in[5];
    const float* g2   = (const float*)d_in[6];
    const float* b2   = (const float*)d_in[7];
    const float* w3   = (const float*)d_in[8];
    const float* g3   = (const float*)d_in[9];
    const float* b3   = (const float*)d_in[10];
    const float* w4   = (const float*)d_in[11];
    const float* g4   = (const float*)d_in[12];
    const float* b4   = (const float*)d_in[13];
    const float* w5   = (const float*)d_in[14];
    const float* g5   = (const float*)d_in[15];
    const float* b5   = (const float*)d_in[16];
    const float* wl1  = (const float*)d_in[17];
    const float* g6   = (const float*)d_in[18];
    const float* b6   = (const float*)d_in[19];
    const float* wl2  = (const float*)d_in[20];
    const float* g7   = (const float*)d_in[21];
    const float* b7   = (const float*)d_in[22];
    const float* wl21 = (const float*)d_in[23];
    const float* wl22 = (const float*)d_in[24];
    const float* g8   = (const float*)d_in[25];
    const float* b8   = (const float*)d_in[26];
    const float* wl3  = (const float*)d_in[27];
    float* out = (float*)d_out;

    char* ws = (char*)d_ws;
    size_t off = 0;
    auto alloc = [&](size_t bytes) { size_t o = off; off += (bytes + 255) & ~size_t(255); return o; };
    int*    idx  = (int*)   (ws + alloc((size_t)NBATCH * NPTS * KNN_K * 4));
    bf16_t* Xbf  = (bf16_t*)(ws + alloc((size_t)NBATCH * NPTS * 128 * 2));
    bf16_t* Wsb  = (bf16_t*)(ws + alloc((size_t)512 * 128 * 2));
    bf16_t* W5b  = (bf16_t*)(ws + alloc((size_t)1024 * 512 * 2));
    float*  Ust  = (float*) (ws + alloc((size_t)NBATCH * HCH * NPTS * 4));
    float*  H    = (float*) (ws + alloc((size_t)NBATCH * HCH * NPTS * 4));
    bf16_t* Hbf  = (bf16_t*)(ws + alloc((size_t)NBATCH * NPTS * HCH * 2));
    float*  p    = (float*) (ws + alloc((size_t)NBATCH * 2048 * 4));
    float*  z1   = (float*) (ws + alloc((size_t)NBATCH * 256 * 4));
    float*  z2   = (float*) (ws + alloc((size_t)NBATCH * 128 * 4));
    float*  t    = (float*) (ws + alloc((size_t)NBATCH * 64 * 4));
    float*  u    = (float*) (ws + alloc((size_t)NBATCH * 32 * 4));

    const int BN = NBATCH * NPTS;

    struct Blk { const float* in; int inStride; int Ci, Cpad, O, hOff;
                 const float* W; const float* g; const float* bia; };
    Blk blk[4] = {
        { x,              6 * NPTS,   6,   32,  64,  0,   w1, g1, b1 },
        { H,              HCH * NPTS, 64,  64,  64,  64,  w2, g2, b2 },
        { H + 64 * NPTS,  HCH * NPTS, 64,  64,  128, 128, w3, g3, b3 },
        { H + 128 * NPTS, HCH * NPTS, 128, 128, 256, 256, w4, g4, b4 },
    };

    for (int i = 0; i < 4; ++i) {
        const Blk& s = blk[i];
        knn_kernel<<<BN, 32, 0, stream>>>(s.in, s.Ci, s.inStride, idx);
        cvt_act_kernel<<<BN, s.Cpad, 0, stream>>>(s.in, s.inStride, s.Ci, s.Cpad, Xbf);
        wstack_kernel<<<2 * s.O, s.Cpad, 0, stream>>>(s.W, s.Ci, s.O, s.Cpad, Wsb);
        gemm_bf16_kernel<<<dim3(NPTS / 64, (2 * s.O) / 16, NBATCH), 32, 0, stream>>>(
            Wsb, Xbf, Ust, s.Cpad);
        gathermax_kernel<<<BN, s.O, 0, stream>>>(Ust, idx, s.g, s.bia, H, s.O, s.hOff);
    }

    // conv5: transpose-convert H -> [b][n][512] bf16, convert W5, fused GEMM+pool
    cvt_act_kernel<<<BN, HCH, 0, stream>>>(H, HCH * NPTS, HCH, HCH, Hbf);
    cvt_plain_kernel<<<(1024 * 512 + 255) / 256, 256, 0, stream>>>(w5, 1024 * 512, W5b);
    conv5_pool_kernel<<<dim3(1024 / 16 / 4, NBATCH), 128, 0, stream>>>(W5b, Hbf, g5, b5, p);

    // MLP head
    dense_kernel<<<(NBATCH * 256 + 127) / 128, 128, 0, stream>>>(p,  wl1,  g6, b6, z1, 2048, 256, 0);
    dense_kernel<<<(NBATCH * 128 + 127) / 128, 128, 0, stream>>>(z1, wl2,  g7, b7, z2, 256, 128, 0);
    dense_kernel<<<(NBATCH * 64  + 127) / 128, 128, 0, stream>>>(z2, wl21, nullptr, nullptr, t, 128, 64, 2);
    dense_kernel<<<(NBATCH * 32  + 127) / 128, 128, 0, stream>>>(t,  wl22, g8, b8, u, 64, 32, 1);
    dense_kernel<<<(NBATCH * 90  + 127) / 128, 128, 0, stream>>>(u,  wl3,  nullptr, nullptr, out, 32, 90, 2);
}
